// GINEncoder_16114717295311
// MI455X (gfx1250) — matrix operand order repaired and checked by reading, not compiled, not run on hardware
//
#include <hip/hip_runtime.h>

#define NNODES 100000
#define NEDGES 1600000
#define DIM    64
#define NGRAPH 512
#define BN_EPS 1e-5f
#define NSLABS ((NNODES + 63) / 64)

typedef __attribute__((ext_vector_type(16))) __bf16 v16bf;
typedef __attribute__((ext_vector_type(8)))  float  v8f;

#define WMMA_BF16(a, b, c) \
  __builtin_amdgcn_wmma_f32_16x16x32_bf16(false, (a), false, (b), (short)0, (c), false, false)

union Frag16 { int4 q[2]; v16bf v; };   // 32B WMMA operand built from 2x b128
union Pack8  { __bf16 h[8]; int4 q; };  // 8 bf16 <-> one 16B LDS transaction

// ---------------------------------------------------------------- utilities

__global__ void zero_f32(float* __restrict__ p, int n4) {
  int t = blockIdx.x * blockDim.x + threadIdx.x;
  if (t < n4) ((float4*)p)[t] = make_float4(0.f, 0.f, 0.f, 0.f);
}

// ------------------------------------------------- edge scatter-add (agg)
// agg[dst] += BN(h[src]); one thread = one (edge, float4 chunk). Coalesced
// 16B gathers feeding native global_atomic_add_f32 into L2-resident agg.

__global__ __launch_bounds__(256)
void scatter_edges(const float* __restrict__ h,
                   const float* __restrict__ sc,   // nullable BN scale [64]
                   const float* __restrict__ sh,   // nullable BN shift [64]
                   const int*   __restrict__ src,
                   const int*   __restrict__ dst,
                   float*       __restrict__ agg) {
  int t = blockIdx.x * blockDim.x + threadIdx.x;
  if (t >= NEDGES * 16) return;
  int e = t >> 4;
  int q = t & 15;
  int s = src[e];
  int d = dst[e];
  float4 v = ((const float4*)(h + (size_t)s * DIM))[q];
  int c = q * 4;
  if (sc) {
    v.x = v.x * sc[c + 0] + sh[c + 0];
    v.y = v.y * sc[c + 1] + sh[c + 1];
    v.z = v.z * sc[c + 2] + sh[c + 2];
    v.w = v.w * sc[c + 3] + sh[c + 3];
  }
  float* ap = agg + (size_t)d * DIM + c;
  unsafeAtomicAdd(ap + 0, v.x);
  unsafeAtomicAdd(ap + 1, v.y);
  unsafeAtomicAdd(ap + 2, v.z);
  unsafeAtomicAdd(ap + 3, v.w);
}

// ----------------------------------------------------- fragment loaders
// A (weights, pre-permuted in LDS): fragment (ct,kc2) = int4 chunks
//   [ (((ct*2+kc2)*2+half)*2+grp)*16 + mm ], grp stride = 16 int4s.
// B (activations, row-major [n][64] bf16): lane (half,n) takes 16
//   K values kc8*8 + {8*half, 16+8*half} + 0..7: int4s  n*8 + kc8 + half (+2).

__device__ inline v16bf wfrag(const int4* __restrict__ w4, int ct, int kc2,
                              int half, int mm) {
  int b0 = (((ct * 2 + kc2) * 2 + half) * 2 + 0) * 16 + mm;
  Frag16 f;
  f.q[0] = w4[b0];
  f.q[1] = w4[b0 + 16];
  return f.v;
}

__device__ inline v16bf bfrag(const int4* __restrict__ a4, int n, int half,
                              int kc8) {
  Frag16 f;
  int b0 = n * 8 + kc8 + half;
  f.q[0] = a4[b0];
  f.q[1] = a4[b0 + 2];
  return f.v;
}

// --------------------------------------------------- fused GIN MLP (WMMA)
// Computes h2 = relu(relu((BN(x)+agg) @ wa + ba) @ wb + bb) for 64-node
// slabs, as h^T = W^T @ x^T so weights are the A operand. 128 thr = 4 waves,
// one 16-node tile each. All LDS traffic is b128-granular.

__global__ __launch_bounds__(128)
void gin_mlp_wmma(const float* __restrict__ xin,
                  const float* __restrict__ sc,   // nullable
                  const float* __restrict__ sh,
                  const float* __restrict__ agg,
                  const float* __restrict__ wa, const float* __restrict__ ba,
                  const float* __restrict__ wb, const float* __restrict__ bb,
                  float* __restrict__ hout) {
  __shared__ __bf16 lwa[4096];      // wa^T in A-fragment order
  __shared__ __bf16 lwb[4096];      // wb^T in A-fragment order
  __shared__ __bf16 act[4][1024];   // per-wave 16x64 input tile, row-major
  __shared__ __bf16 mid[4][1024];   // per-wave 16x64 hidden tile, row-major

  const int tid  = threadIdx.x;
  const int wave = tid >> 5;
  const int lane = tid & 31;
  const int half = lane >> 4;
  const int nn   = lane & 15;

  // ---- stage both weight matrices to LDS, permuted to A-fragment order.
  // src element wa[k*64+m]; dest chunk decoded from cidx.
  for (int cidx = tid; cidx < 512; cidx += 128) {
    int mm  = cidx & 15;
    int grp = (cidx >> 4) & 1;
    int hf  = (cidx >> 5) & 1;
    int kc  = (cidx >> 6) & 1;
    int ct  = cidx >> 7;
    int k0  = kc * 32 + grp * 16 + hf * 8;
    int m   = ct * 16 + mm;
    Pack8 pa, pb;
#pragma unroll
    for (int j = 0; j < 8; ++j) {
      pa.h[j] = (__bf16)wa[(k0 + j) * 64 + m];
      pb.h[j] = (__bf16)wb[(k0 + j) * 64 + m];
    }
    ((int4*)lwa)[cidx] = pa.q;
    ((int4*)lwb)[cidx] = pb.q;
  }
  __syncthreads();

  const int4* wa4 = (const int4*)lwa;
  const int4* wb4 = (const int4*)lwb;
  const int4* a4  = (const int4*)act[wave];
  const int4* m4  = (const int4*)mid[wave];

  for (int slab = blockIdx.x; slab < NSLABS; slab += gridDim.x) {
    const int rowBase = slab * 64 + wave * 16;

    // ---- stage activation tile: BN(x) + agg -> bf16, one b128 per 8 vals
    for (int i = lane; i < 128; i += 32) {
      int m   = i >> 3;
      int c8  = (i & 7) * 8;
      int row = rowBase + m;
      float4 x0 = make_float4(0.f, 0.f, 0.f, 0.f), x1 = x0, g0 = x0, g1 = x0;
      if (row < NNODES) {
        const float4* xp = (const float4*)(xin + (size_t)row * DIM + c8);
        const float4* gp = (const float4*)(agg + (size_t)row * DIM + c8);
        x0 = xp[0]; x1 = xp[1];
        g0 = gp[0]; g1 = gp[1];
      }
      float xf[8] = {x0.x, x0.y, x0.z, x0.w, x1.x, x1.y, x1.z, x1.w};
      float gf[8] = {g0.x, g0.y, g0.z, g0.w, g1.x, g1.y, g1.z, g1.w};
      Pack8 p;
      if (sc) {
#pragma unroll
        for (int j = 0; j < 8; ++j)
          p.h[j] = (__bf16)(xf[j] * sc[c8 + j] + sh[c8 + j] + gf[j]);
      } else {
#pragma unroll
        for (int j = 0; j < 8; ++j)
          p.h[j] = (__bf16)(xf[j] + gf[j]);
      }
      ((int4*)act[wave])[i] = p.q;
    }
    // act/mid tiles are wave-private: lockstep wave32 + compiler dscnt waits
    // give ordering without a block barrier.

    // ---- GEMM 1: h1^T = wa^T @ act^T ; bias+relu ; pack to mid (1 b128/ct)
    v16bf bact0 = bfrag(a4, nn, half, 0);
    v16bf bact1 = bfrag(a4, nn, half, 4);
#pragma unroll
    for (int ct = 0; ct < 4; ++ct) {
      v8f c = {};
      c = WMMA_BF16(wfrag(wa4, ct, 0, half, nn), bact0, c);
      c = WMMA_BF16(wfrag(wa4, ct, 1, half, nn), bact1, c);
      const float4* bp = (const float4*)(ba + ct * 16 + half * 8);
      float4 bl = bp[0], bh = bp[1];
      float bias[8] = {bl.x, bl.y, bl.z, bl.w, bh.x, bh.y, bh.z, bh.w};
      Pack8 p;
#pragma unroll
      for (int r = 0; r < 8; ++r) {
        float v = c[r] + bias[r];
        p.h[r] = (__bf16)(v > 0.f ? v : 0.f);
      }
      // lane (half,nn) holds channels ct*16+8*half+r of node nn: contiguous
      ((int4*)mid[wave])[nn * 8 + ct * 2 + half] = p.q;
    }

    // ---- GEMM 2: h2^T = wb^T @ h1^T ; bias+relu ; store 2x b128 per ct
    v16bf bmid0 = bfrag(m4, nn, half, 0);
    v16bf bmid1 = bfrag(m4, nn, half, 4);
    const int node = rowBase + nn;
#pragma unroll
    for (int ct = 0; ct < 4; ++ct) {
      v8f c = {};
      c = WMMA_BF16(wfrag(wb4, ct, 0, half, nn), bmid0, c);
      c = WMMA_BF16(wfrag(wb4, ct, 1, half, nn), bmid1, c);
      const float4* bp = (const float4*)(bb + ct * 16 + half * 8);
      float4 bl = bp[0], bh = bp[1];
      float bias[8] = {bl.x, bl.y, bl.z, bl.w, bh.x, bh.y, bh.z, bh.w};
      float v[8];
#pragma unroll
      for (int r = 0; r < 8; ++r) {
        float t = c[r] + bias[r];
        v[r] = t > 0.f ? t : 0.f;
      }
      if (node < NNODES) {
        float4* op = (float4*)(hout + (size_t)node * DIM + ct * 16 + half * 8);
        op[0] = make_float4(v[0], v[1], v[2], v[3]);
        op[1] = make_float4(v[4], v[5], v[6], v[7]);
      }
    }
  }
}

// ------------------------------------------- per-column sum / sumsq (BN)

__global__ __launch_bounds__(256)
void col_stats(const float* __restrict__ h, float* __restrict__ stats) {
  __shared__ float ssum[256];
  __shared__ float ssq[256];
  int col = threadIdx.x & 63;
  int grp = threadIdx.x >> 6;  // 0..3
  float s = 0.f, q = 0.f;
  for (int row = blockIdx.x * 4 + grp; row < NNODES; row += gridDim.x * 4) {
    float v = h[(size_t)row * DIM + col];
    s += v;
    q += v * v;
  }
  ssum[threadIdx.x] = s;
  ssq[threadIdx.x]  = q;
  __syncthreads();
  if (threadIdx.x < 64) {
    s = ssum[threadIdx.x] + ssum[threadIdx.x + 64] + ssum[threadIdx.x + 128] +
        ssum[threadIdx.x + 192];
    q = ssq[threadIdx.x] + ssq[threadIdx.x + 64] + ssq[threadIdx.x + 128] +
        ssq[threadIdx.x + 192];
    unsafeAtomicAdd(&stats[col], s);
    unsafeAtomicAdd(&stats[DIM + col], q);
  }
}

// ------------------------------------- BN: stats -> per-column affine

__global__ void bn_params(const float* __restrict__ stats,
                          const float* __restrict__ gamma,
                          const float* __restrict__ beta,
                          float* __restrict__ scale,
                          float* __restrict__ shift) {
  int c = threadIdx.x;
  if (c >= DIM) return;
  float invN = 1.0f / (float)NNODES;
  float mean = stats[c] * invN;
  float var  = stats[DIM + c] * invN - mean * mean;
  float s    = gamma[c] * rsqrtf(var + BN_EPS);
  scale[c] = s;
  shift[c] = beta[c] - mean * s;
}

// --------------------------------------------- global add pool over graphs

__global__ __launch_bounds__(256)
void pool_graphs(const float* __restrict__ h,
                 const float* __restrict__ sc,
                 const float* __restrict__ sh,
                 const int*   __restrict__ batch,
                 float*       __restrict__ out) {
  int t = blockIdx.x * blockDim.x + threadIdx.x;
  if (t >= NNODES * 16) return;
  int n = t >> 4;
  int q = t & 15;
  int c = q * 4;
  int b = batch[n];
  float4 v = ((const float4*)(h + (size_t)n * DIM))[q];
  float* op = out + (size_t)b * DIM + c;
  unsafeAtomicAdd(op + 0, v.x * sc[c + 0] + sh[c + 0]);
  unsafeAtomicAdd(op + 1, v.y * sc[c + 1] + sh[c + 1]);
  unsafeAtomicAdd(op + 2, v.z * sc[c + 2] + sh[c + 2]);
  unsafeAtomicAdd(op + 3, v.w * sc[c + 3] + sh[c + 3]);
}

// -------------------------------------------------------------------- driver

extern "C" void kernel_launch(void* const* d_in, const int* in_sizes, int n_in,
                              void* d_out, int out_size, void* d_ws, size_t ws_size,
                              hipStream_t stream) {
  const float* x     = (const float*)d_in[0];
  const int*   ei    = (const int*)d_in[1];
  const int*   src   = ei;
  const int*   dst   = ei + NEDGES;
  const int*   batch = (const int*)d_in[2];

  // layer params: wa, ba, wb, bb, gamma, beta
  const float* P[3][6];
  for (int l = 0; l < 3; ++l)
    for (int j = 0; j < 6; ++j)
      P[l][j] = (const float*)d_in[3 + l * 6 + j];

  // workspace layout
  float* agg   = (float*)d_ws;                       // N*64
  float* hA    = agg   + (size_t)NNODES * DIM;       // N*64
  float* hB    = hA    + (size_t)NNODES * DIM;       // N*64
  float* stats = hB    + (size_t)NNODES * DIM;       // 3 * 128
  float* scsh  = stats + 3 * 128;                    // 3 * 128

  const size_t ND = (size_t)NNODES * DIM;

  const float* hin = x;
  const float* sc  = nullptr;
  const float* sh  = nullptr;
  float* houts[3] = {hA, hB, hA};

  const int edgeThreads = NEDGES * 16;

  for (int L = 0; L < 3; ++L) {
    float* st = stats + L * 128;
    zero_f32<<<(int)((ND / 4 + 255) / 256), 256, 0, stream>>>(agg, (int)(ND / 4));
    zero_f32<<<1, 32, 0, stream>>>(st, 128 / 4);

    scatter_edges<<<(edgeThreads + 255) / 256, 256, 0, stream>>>(
        hin, sc, sh, src, dst, agg);

    gin_mlp_wmma<<<512, 128, 0, stream>>>(
        hin, sc, sh, agg,
        P[L][0], P[L][1], P[L][2], P[L][3],
        houts[L]);

    col_stats<<<512, 256, 0, stream>>>(houts[L], st);

    bn_params<<<1, 64, 0, stream>>>(st, P[L][4], P[L][5],
                                    scsh + L * 128, scsh + L * 128 + 64);

    hin = houts[L];
    sc  = scsh + L * 128;
    sh  = sc + 64;
  }

  zero_f32<<<(NGRAPH * DIM / 4 + 255) / 256, 256, 0, stream>>>((float*)d_out,
                                                               NGRAPH * DIM / 4);
  pool_graphs<<<(NNODES * 16 + 255) / 256, 256, 0, stream>>>(
      hin, sc, sh, batch, (float*)d_out);
}
